// GPTQLinearOnnxExact_90228672954454
// MI455X (gfx1250) — compile-verified, matrix operand order checked by reading, not run on hardware
//
#include <hip/hip_runtime.h>

// ---------------------------------------------------------------------------
// GPTQ 4-bit dequant + GEMM for MI455X (gfx1250), wave32 + WMMA f16->f32.
//   out[t, n] = sum_k x[t,k] * scales[g(k),n] * (q[k,n] - zp[g(k),n]) + bias[n]
// M = 4096 tokens, N = 4096, K = 4096, group size 128.
// Double-buffered LDS ping-pong, BK=64 (2 WMMA K-steps per barrier),
// packed ds_store_b64 commits for both A and B tiles.
// ---------------------------------------------------------------------------

typedef __attribute__((ext_vector_type(16), aligned(16))) _Float16 v16h;
typedef __attribute__((ext_vector_type(8)))  float    v8f;
typedef __attribute__((ext_vector_type(4)))  float    f32x4;
typedef __attribute__((ext_vector_type(4)))  int      i32x4;
typedef __attribute__((ext_vector_type(4), aligned(8))) _Float16 h16x4;

#define IN_F   4096
#define OUT_F  4096
#define BM     128
#define BN     128
#define BK     64
#define PITCH  72   // halves per LDS row/col (64 + 8 pad), 144 B (16-B aligned)

__global__ __launch_bounds__(256)
void gptq_wmma_kernel(const float* __restrict__ x,
                      const int*   __restrict__ qw,
                      const float* __restrict__ scales,
                      const int*   __restrict__ zp,
                      const int*   __restrict__ gidx,
                      const float* __restrict__ bias,
                      float*       __restrict__ out)
{
    // ping-pong buffers: A row-major [row][k], B col-major [col][k]; 72 KB total
    __shared__ _Float16 lA[2][BM * PITCH];
    __shared__ _Float16 lB[2][BN * PITCH];

    const int t     = threadIdx.x;
    const int lane  = t & 31;
    const int wave  = t >> 5;        // 0..7
    const int waveM = wave >> 1;     // 0..3  -> 32 rows each
    const int waveN = wave & 1;      // 0..1  -> 64 cols each
    const int rowBase = blockIdx.y * BM;
    const int colBase = blockIdx.x * BN;

    const int l15  = lane & 15;
    const int half = lane >> 4;      // 0/1 -> low/high 16 of the 32-wide K step

    // A staging map: 128 rows x 16 float4 = 2048 tasks / 256 thr = 8 each
    const int aF4   = t & 15;        // float4 slot within 64-wide K row
    const int aRow0 = t >> 4;        // 0..15, step +16
    // B staging map: 4 output columns (nl..nl+3), 8 CONSECUTIVE k rows kB0..kB0+7
    const int nl    = (t & 31) * 4;  // 0..124, local output column
    const int kB0   = (t >> 5) * 8;  // 0,8,...,56

    v8f acc[2][4];
#pragma unroll
    for (int tm = 0; tm < 2; ++tm)
#pragma unroll
        for (int tn = 0; tn < 4; ++tn)
            acc[tm][tn] = (v8f)(0.0f);

    // ---------------- prologue: stage tile k0=0 into buffer 0 ----------------
    {
#pragma unroll
        for (int i = 0; i < 8; ++i) {
            const int r = aRow0 + i * 16;
            f32x4 v = *(const f32x4*)&x[(size_t)(rowBase + r) * IN_F + aF4 * 4];
            h16x4 h;
            h.x = (_Float16)v.x; h.y = (_Float16)v.y;
            h.z = (_Float16)v.z; h.w = (_Float16)v.w;
            *(h16x4*)&lA[0][r * PITCH + aF4 * 4] = h;
        }
        const int gi = gidx[0];
        const size_t gOff = (size_t)gi * OUT_F + colBase + nl;
        f32x4 sc = *(const f32x4*)&scales[gOff];
        i32x4 z  = *(const i32x4*)&zp[gOff];
        i32x4 q[8];
#pragma unroll
        for (int i = 0; i < 8; ++i)
            q[i] = *(const i32x4*)&qw[(size_t)(kB0 + i) * OUT_F + colBase + nl];
#pragma unroll
        for (int j = 0; j < 4; ++j) {
            const float scj = sc[j]; const int zj = z[j];
            h16x4 lo, hi;
            lo.x = (_Float16)(scj * (float)(q[0][j] - zj));
            lo.y = (_Float16)(scj * (float)(q[1][j] - zj));
            lo.z = (_Float16)(scj * (float)(q[2][j] - zj));
            lo.w = (_Float16)(scj * (float)(q[3][j] - zj));
            hi.x = (_Float16)(scj * (float)(q[4][j] - zj));
            hi.y = (_Float16)(scj * (float)(q[5][j] - zj));
            hi.z = (_Float16)(scj * (float)(q[6][j] - zj));
            hi.w = (_Float16)(scj * (float)(q[7][j] - zj));
            _Float16* base = &lB[0][(nl + j) * PITCH + kB0];
            *(h16x4*)(base)     = lo;
            *(h16x4*)(base + 4) = hi;
        }
    }
    __syncthreads();

    // ---------------- main loop: issue(next) -> compute(cur) -> commit(next) ----
    int cur = 0;
    for (int k0 = 0; k0 < IN_F; k0 += BK) {
        const int  kNext = k0 + BK;
        const bool more  = kNext < IN_F;   // uniform
        const int  nxt   = cur ^ 1;

        // -- issue next tile's global loads (in flight during compute) --
        f32x4 aReg[8]; i32x4 qReg[8]; f32x4 sc; i32x4 z;
        if (more) {
#pragma unroll
            for (int i = 0; i < 8; ++i) {
                const int r = aRow0 + i * 16;
                aReg[i] = *(const f32x4*)&x[(size_t)(rowBase + r) * IN_F + kNext + aF4 * 4];
            }
            const int gi = gidx[kNext];   // group constant over 64-wide tile
            const size_t gOff = (size_t)gi * OUT_F + colBase + nl;
            sc = *(const f32x4*)&scales[gOff];
            z  = *(const i32x4*)&zp[gOff];
#pragma unroll
            for (int i = 0; i < 8; ++i)
                qReg[i] = *(const i32x4*)&qw[(size_t)(kNext + kB0 + i) * OUT_F + colBase + nl];
        }

        // -- compute: 2 WMMA K-steps from buffer `cur` --
        // 16-bit A fragment, this kernel's k order: lane l<16 holds row M=l, K=0..15 contiguous;
        // lanes 16..31 hold K=16..31. B symmetric (stored col-major).
#pragma unroll
        for (int s = 0; s < 2; ++s) {
            const int koff = s * 32 + half * 16;
            v16h aF[2], bF[4];
#pragma unroll
            for (int tm = 0; tm < 2; ++tm)
                aF[tm] = *(const v16h*)&lA[cur][(waveM * 32 + tm * 16 + l15) * PITCH + koff];
#pragma unroll
            for (int tn = 0; tn < 4; ++tn)
                bF[tn] = *(const v16h*)&lB[cur][(waveN * 64 + tn * 16 + l15) * PITCH + koff];
#pragma unroll
            for (int tm = 0; tm < 2; ++tm)
#pragma unroll
                for (int tn = 0; tn < 4; ++tn)
                    acc[tm][tn] = __builtin_amdgcn_wmma_f32_16x16x32_f16(
                        /*neg_a=*/false, aF[tm],
                        /*neg_b=*/false, bF[tn],
                        /*c_mod=*/(short)0, acc[tm][tn],
                        /*reuse_a=*/false, /*reuse_b=*/false);
        }

        // -- commit staged tile to buffer `nxt` (waits loads here, after WMMAs) --
        if (more) {
#pragma unroll
            for (int i = 0; i < 8; ++i) {
                const int r = aRow0 + i * 16;
                h16x4 h;
                h.x = (_Float16)aReg[i].x; h.y = (_Float16)aReg[i].y;
                h.z = (_Float16)aReg[i].z; h.w = (_Float16)aReg[i].w;
                *(h16x4*)&lA[nxt][r * PITCH + aF4 * 4] = h;
            }
#pragma unroll
            for (int j = 0; j < 4; ++j) {
                const float scj = sc[j]; const int zj = z[j];
                h16x4 lo, hi;
                lo.x = (_Float16)(scj * (float)(qReg[0][j] - zj));
                lo.y = (_Float16)(scj * (float)(qReg[1][j] - zj));
                lo.z = (_Float16)(scj * (float)(qReg[2][j] - zj));
                lo.w = (_Float16)(scj * (float)(qReg[3][j] - zj));
                hi.x = (_Float16)(scj * (float)(qReg[4][j] - zj));
                hi.y = (_Float16)(scj * (float)(qReg[5][j] - zj));
                hi.z = (_Float16)(scj * (float)(qReg[6][j] - zj));
                hi.w = (_Float16)(scj * (float)(qReg[7][j] - zj));
                _Float16* base = &lB[nxt][(nl + j) * PITCH + kB0];
                *(h16x4*)(base)     = lo;
                *(h16x4*)(base + 4) = hi;
            }
        }

        __syncthreads();   // one barrier per tile: publishes nxt, retires cur
        cur = nxt;
    }

    // ---- epilogue: bias add + store (C/D layout: VGPR r -> M = r + 8*half) ----
#pragma unroll
    for (int tm = 0; tm < 2; ++tm) {
        const int gr0 = rowBase + waveM * 32 + tm * 16 + half * 8;
#pragma unroll
        for (int tn = 0; tn < 4; ++tn) {
            const int gc = colBase + waveN * 64 + tn * 16 + l15;
            const float b = bias[gc];
#pragma unroll
            for (int r = 0; r < 8; ++r)
                out[(size_t)(gr0 + r) * OUT_F + gc] = acc[tm][tn][r] + b;
        }
    }
}

extern "C" void kernel_launch(void* const* d_in, const int* in_sizes, int n_in,
                              void* d_out, int out_size, void* d_ws, size_t ws_size,
                              hipStream_t stream) {
    const float* x      = (const float*)d_in[0];
    const int*   qw     = (const int*)d_in[1];
    const float* scales = (const float*)d_in[2];
    const int*   zp     = (const int*)d_in[3];
    const int*   gidx   = (const int*)d_in[4];
    const float* bias   = (const float*)d_in[5];
    float*       out    = (float*)d_out;

    const int tokens = in_sizes[0] / IN_F;       // 2*2048 = 4096
    dim3 grid(OUT_F / BN, tokens / BM);          // 32 x 32 blocks
    gptq_wmma_kernel<<<grid, 256, 0, stream>>>(x, qw, scales, zp, gidx, bias, out);
}
